// Generator_82042465288579
// MI455X (gfx1250) — matrix operand order repaired and checked by reading, not compiled, not run on hardware
//
#include <hip/hip_runtime.h>
#include <hip/hip_bf16.h>

// ---------------- problem constants ----------------
#define HID    64
#define NL     8
#define TMAX   64
#define GATES  256                 // 4*HID
#define VOCAB  50257
#define NBLOG  ((VOCAB + 255) / 256)

// workspace layout (bytes)
#define WSO_SEQ    0            // 64 ints
#define WSO_DONE   256          // 1 int
#define WSO_COMB   512          // 512 floats  -> ends 2560
#define WSO_PVAL   2560         // 256 floats  -> ends 3584
#define WSO_PIDX   3584         // 256 ints    -> ends 4608
#define WSO_HCHK   4608         // 512 floats  -> ends 6656
#define WSO_CCHK   6656         // 512 floats  -> ends 8704
#define WSO_BIAS   8704         // 2048 floats -> ends 16896
#define WSO_WPACK  16896        // 262144 halves -> ends 541184
#define WS_NEED_CHK   ((size_t)8704)
#define WS_NEED_FULL  ((size_t)(16896 + NL*16*32*64*2))

typedef __attribute__((ext_vector_type(16))) _Float16 v16h;
typedef __attribute__((ext_vector_type(8)))  _Float16 v8h;
typedef __attribute__((ext_vector_type(8)))  float    v8f;

__device__ __forceinline__ float fast_sigmoid(float x) {
  // v_exp_f32 + v_rcp_f32 (no IEEE-div expansion on the serial path)
  return __builtin_amdgcn_rcpf(1.0f + __expf(-x));
}
__device__ __forceinline__ float fast_tanh(float x) {
  return 2.0f * __builtin_amdgcn_rcpf(1.0f + __expf(-2.0f * x)) - 1.0f;
}

// ============================================================
// One-time prep: pack LSTM weights into the exact per-lane WMMA
// B-operand f16 layout  wpack[l][wave][lane][64], and fold biases.
// ============================================================
__global__ __launch_bounds__(256) void prep_weights_kernel(
    const float* __restrict__ Wih, const float* __restrict__ Whh,
    _Float16* __restrict__ wpack)
{
  int idx = blockIdx.x * 256 + threadIdx.x;       // over NL*16*32*64
  if (idx >= NL * 16 * 32 * 64) return;
  int e  = idx & 63;
  int L  = (idx >> 6)  & 31;
  int w  = (idx >> 11) & 15;
  int l  = idx >> 15;
  int c  = e >> 4, el = e & 15;
  int kk = 32 * c + ((L < 16) ? 0 : 16) + el;     // K over concat [x | h]
  int o  = w * 16 + (L & 15);                     // gate row (N)
  float v = (kk < HID) ? Wih[(size_t)l * GATES * HID + o * HID + kk]
                       : Whh[(size_t)l * GATES * HID + o * HID + (kk - HID)];
  wpack[idx] = (_Float16)v;
}

__global__ __launch_bounds__(256) void prep_bias_kernel(
    const float* __restrict__ b_ih, const float* __restrict__ b_hh,
    float* __restrict__ biasp)
{
  int idx = blockIdx.x * 256 + threadIdx.x;
  if (idx < NL * GATES) biasp[idx] = b_ih[idx] + b_hh[idx];
}

// ============================================================
// Kernel A: 8-layer LSTM over the *suffix* t = istart..63 only
// (prefix activations are unchanged between decode steps; per-layer
// (h,c) checkpoints live in ws). 512 threads = 16 wave32s; wave w
// computes gates [16w,16w+16) with v_wmma_f32_16x16x32_f16, weights
// VGPR-resident per layer.
// ============================================================
__global__ __launch_bounds__(512) void lstm_step_kernel(
    const float* __restrict__ emb,   const float* __restrict__ Wih,
    const float* __restrict__ Whh,   const float* __restrict__ b_ih,
    const float* __restrict__ b_hh,  const float* __restrict__ fc_w,
    const float* __restrict__ fc_b,  const float* __restrict__ img,
    const int*   __restrict__ seq,   float* __restrict__ comb,
    const _Float16* __restrict__ wpack, const float* __restrict__ biasp,
    float* __restrict__ hchk, float* __restrict__ cchk, int istart)
{
  __shared__ __align__(16) _Float16 hseq[TMAX][HID];  // layer in/out seq (f16)
  __shared__ __align__(16) _Float16 hcur[HID];        // recurrent h (f16)
  __shared__ float cstate[HID];
  __shared__ float gates[GATES];
  __shared__ float hlast[HID];                        // top-layer h (f32)
  __shared__ int   seq_s[TMAX];

  const int tid  = threadIdx.x;
  const int wave = tid >> 5;
  const int lane = tid & 31;
  const int ntok = TMAX - istart;

  if (tid < ntok) seq_s[istart + tid] = seq[istart + tid];
  __syncthreads();

  // Embedding lookup for suffix tokens -> hseq (layer-0 inputs), f16
  for (int e = tid; e < ntok * HID; e += 512) {
    int t = istart + (e >> 6), k = e & 63;
    hseq[t][k] = (_Float16)emb[(size_t)seq_s[t] * HID + k];
  }
  __syncthreads();

  // Per-lane WMMA operand maps (contiguous K order used by this code, wave32):
  //  B (32x16): lanes 0-15 K=0..15 (elem e -> K=e), lanes 16-31 K=16..31; N = lane&15
  //  A (16x32): lanes 0-15: elem e -> K=e; lanes 16-31: K=16+e (matches B)
  const int n     = lane & 15;
  const int kboff = (lane < 16) ? 0  : 16;
  const int aoff  = (lane < 16) ? 0  : 16;
  const int aoff2 = (lane < 16) ? 8  : 24;
  const int o     = wave * 16 + n;

  for (int l = 0; l < NL; ++l) {
    // ---- layer weights -> VGPRs (f16 B tiles) ----
    v16h Bw[4];
    float bias;
    if (wpack) {
      const v16h* wp =
          (const v16h*)(wpack + ((((size_t)l * 16 + wave) * 32 + lane) << 6));
#pragma unroll
      for (int c = 0; c < 4; ++c) Bw[c] = wp[c];
      bias = biasp[l * GATES + o];
    } else {
      const float* Wi = Wih + (size_t)l * GATES * HID;
      const float* Wh = Whh + (size_t)l * GATES * HID;
#pragma unroll
      for (int c = 0; c < 4; ++c) {
#pragma unroll
        for (int e = 0; e < 16; ++e) {
          int kk = 32 * c + kboff + e;
          float w = (kk < HID) ? Wi[o * HID + kk] : Wh[o * HID + (kk - HID)];
          Bw[c][e] = (_Float16)w;
        }
      }
      bias = b_ih[l * GATES + o] + b_hh[l * GATES + o];
    }

    // ---- (h,c) entering t = istart ----
    if (tid < HID) {
      if (istart == 0) {
        cstate[tid] = 0.0f;
        hcur[tid]   = (_Float16)0.0f;
      } else {
        cstate[tid] = cchk[l * HID + tid];
        hcur[tid]   = (_Float16)hchk[l * HID + tid];
      }
    }
    __syncthreads();

    for (int t = istart; t < TMAX; ++t) {
      // ---- A = input vector replicated over M rows; 4 chained WMMAs ----
      v8f acc = {};
#pragma unroll
      for (int c = 0; c < 4; ++c) {
        const _Float16* src  = (c < 2) ? &hseq[t][0] : &hcur[0];
        const int       base = 32 * c - ((c < 2) ? 0 : HID);
        v8h lo = *(const v8h*)(src + base + aoff);
        v8h hi = *(const v8h*)(src + base + aoff2);
        v16h Ac;
#pragma unroll
        for (int e = 0; e < 8; ++e) { Ac[e] = lo[e]; Ac[8 + e] = hi[e]; }
        acc = __builtin_amdgcn_wmma_f32_16x16x32_f16(
            false, Ac, false, Bw[c], (short)0, acc, false, false);
      }
      if (lane < 16) gates[wave * 16 + lane] = acc[0] + bias;
      __syncthreads();

      // ---- LSTM cell (torch gate order i,f,g,o) ----
      if (tid < HID) {
        float i_ = fast_sigmoid(gates[tid]);
        float f_ = fast_sigmoid(gates[HID + tid]);
        float g_ = fast_tanh   (gates[2 * HID + tid]);
        float o_ = fast_sigmoid(gates[3 * HID + tid]);
        float c  = f_ * cstate[tid] + i_ * g_;
        cstate[tid] = c;
        float h = o_ * fast_tanh(c);
        _Float16 hh = (_Float16)h;
        hcur[tid]    = hh;
        hseq[t][tid] = hh;                 // next layer's input at time t
        if (t == istart && hchk) {         // checkpoint = state entering t+1
          hchk[l * HID + tid] = h;
          cchk[l * HID + tid] = c;
        }
        if (l == NL - 1) hlast[tid] = h;   // last write (t=63) is the one used
      }
      __syncthreads();
    }
  }

  // ---- txt = fc_w @ h_last + fc_b ; comb = [img | txt] ----
  if (tid < 256) {
    const float* wr = fc_w + tid * HID;
    float s = fc_b[tid];
#pragma unroll 8
    for (int k = 0; k < HID; ++k) s += wr[k] * hlast[k];
    comb[256 + tid] = s;
    comb[tid]       = img[tid];
  }
}

// ============================================================
// Kernel B: logits = fcc_w @ comb + fcc_b, per-block argmax.
// 103 MB fcc_w is L2-resident (192 MB L2); full-chip sweep.
// ============================================================
__global__ __launch_bounds__(256) void logits_argmax_kernel(
    const float* __restrict__ fcc_w, const float* __restrict__ fcc_b,
    const float* __restrict__ comb,
    float* __restrict__ part_val, int* __restrict__ part_idx)
{
  __shared__ __align__(16) float combs[512];
  __shared__ float svals[256];
  __shared__ int   sidx[256];
  const int tid = threadIdx.x;
  combs[tid]       = comb[tid];
  combs[256 + tid] = comb[256 + tid];
  __syncthreads();

  const int r = blockIdx.x * 256 + tid;
  float best = -1e30f;
  int   bidx = 0x7fffffff;
  if (r < VOCAB) {
    const float4* wr = (const float4*)(fcc_w + (size_t)r * 512);
    const float4* cc = (const float4*)combs;
    float s = fcc_b[r];
#pragma unroll 4
    for (int k = 0; k < 128; ++k) {
      float4 w = wr[k];
      float4 c = cc[k];
      s += w.x * c.x + w.y * c.y + w.z * c.z + w.w * c.w;
    }
    best = s; bidx = r;
  }
  svals[tid] = best; sidx[tid] = bidx;
  __syncthreads();
  for (int off = 128; off > 0; off >>= 1) {
    if (tid < off) {
      float v2 = svals[tid + off]; int i2 = sidx[tid + off];
      if (v2 > svals[tid] || (v2 == svals[tid] && i2 < sidx[tid])) {
        svals[tid] = v2; sidx[tid] = i2;
      }
    }
    __syncthreads();
  }
  if (tid == 0) { part_val[blockIdx.x] = svals[0]; part_idx[blockIdx.x] = sidx[0]; }
}

// ============================================================
// Kernel C: final argmax reduction + EOS masking + token write.
// ============================================================
__global__ __launch_bounds__(256) void finalize_kernel(
    const float* __restrict__ part_val, const int* __restrict__ part_idx,
    int nparts, int* seq, int* done, int* out_seq, int step)
{
  __shared__ float sv[256];
  __shared__ int   si[256];
  const int tid = threadIdx.x;
  float best = -1e30f; int bidx = 0x7fffffff;
  for (int p = tid; p < nparts; p += 256) {
    float v = part_val[p]; int i = part_idx[p];
    if (v > best || (v == best && i < bidx)) { best = v; bidx = i; }
  }
  sv[tid] = best; si[tid] = bidx;
  __syncthreads();
  for (int off = 128; off > 0; off >>= 1) {
    if (tid < off) {
      float v2 = sv[tid + off]; int i2 = si[tid + off];
      if (v2 > sv[tid] || (v2 == sv[tid] && i2 < si[tid])) { sv[tid] = v2; si[tid] = i2; }
    }
    __syncthreads();
  }
  if (tid == 0) {
    int d = done[0];
    int w = d ? 0 : si[0];
    seq[step + 1]     = w;
    out_seq[step + 1] = w;
    done[0] = d | (w == 2);   // EOS token id 2
  }
}

__global__ void init_kernel(int* seq, int* done, int* out_seq) {
  const int tid = threadIdx.x;
  if (tid < TMAX) { int v = (tid == 0) ? 1 : 0; seq[tid] = v; out_seq[tid] = v; }
  if (tid == 0) done[0] = 0;
}

// ============================================================
extern "C" void kernel_launch(void* const* d_in, const int* in_sizes, int n_in,
                              void* d_out, int out_size, void* d_ws, size_t ws_size,
                              hipStream_t stream) {
  const float* img   = (const float*)d_in[0];
  const float* emb   = (const float*)d_in[1];
  const float* Wih   = (const float*)d_in[2];
  const float* Whh   = (const float*)d_in[3];
  const float* b_ih  = (const float*)d_in[4];
  const float* b_hh  = (const float*)d_in[5];
  const float* fc_w  = (const float*)d_in[6];
  const float* fc_b  = (const float*)d_in[7];
  const float* fcc_w = (const float*)d_in[8];
  const float* fcc_b = (const float*)d_in[9];
  int* out_seq = (int*)d_out;

  char*  ws       = (char*)d_ws;
  int*   seq      = (int*)(ws + WSO_SEQ);
  int*   done     = (int*)(ws + WSO_DONE);
  float* comb     = (float*)(ws + WSO_COMB);
  float* part_val = (float*)(ws + WSO_PVAL);
  int*   part_idx = (int*)(ws + WSO_PIDX);
  float* hchk     = (float*)(ws + WSO_HCHK);
  float* cchk     = (float*)(ws + WSO_CCHK);
  float* biasp    = (float*)(ws + WSO_BIAS);
  _Float16* wpack = (_Float16*)(ws + WSO_WPACK);

  const bool use_chk  = ws_size >= WS_NEED_CHK;
  const bool use_pack = ws_size >= WS_NEED_FULL;

  init_kernel<<<1, 64, 0, stream>>>(seq, done, out_seq);
  if (use_pack) {
    prep_weights_kernel<<<(NL * 16 * 32 * 64 + 255) / 256, 256, 0, stream>>>(
        Wih, Whh, wpack);
    prep_bias_kernel<<<(NL * GATES + 255) / 256, 256, 0, stream>>>(b_ih, b_hh,
                                                                   biasp);
  }

  // Fixed 63-step greedy decode, host-unrolled (graph-capture friendly;
  // the reference scan has no early exit, only done-masking). Step i only
  // recomputes LSTM timesteps t >= i (prefix is unchanged).
  for (int i = 0; i < TMAX - 1; ++i) {
    const int istart = use_chk ? i : 0;
    lstm_step_kernel<<<1, 512, 0, stream>>>(
        emb, Wih, Whh, b_ih, b_hh, fc_w, fc_b, img, seq, comb,
        use_pack ? wpack : (const _Float16*)nullptr,
        use_pack ? biasp : (const float*)nullptr,
        use_chk ? hchk : (float*)nullptr, use_chk ? cchk : (float*)nullptr,
        istart);
    logits_argmax_kernel<<<NBLOG, 256, 0, stream>>>(fcc_w, fcc_b, comb,
                                                    part_val, part_idx);
    finalize_kernel<<<1, 256, 0, stream>>>(part_val, part_idx, NBLOG,
                                           seq, done, out_seq, i);
  }
}
